// TreeModelAttMD_65429531787913
// MI455X (gfx1250) — compile-verified, matrix operand order checked by reading, not run on hardware
//
#include <hip/hip_runtime.h>
#include <hip/hip_bf16.h>

#define B_   64
#define T_   48
#define H_   512
#define E_   512
#define ATT_ 196
#define AF_  2048
#define FC_  2048
#define V_   10000
#define VPAD_ 10048   // V_ rounded up to a multiple of 64 (4 N-tiles)

typedef __attribute__((ext_vector_type(16))) _Float16 v16h;
typedef __attribute__((ext_vector_type(8)))  float    v8f;

// ---------------------------------------------------------------------------
// WMMA fragment loaders (wave32, 16x16x32 f16 tile).
// A fragment: lane L holds row m = L&15; K order per CDNA5 ISA 16-bit A layout:
//   halves 0..7 -> K = ((L>>4)*8) + h ; halves 8..15 -> K = 16 + ((L>>4)*8) + (h-8)
// B fragment (for C = A * W^T with row-major W[N][K]): lane L holds col n = L&15,
// reading row n of W with the identical K striping.
// ---------------------------------------------------------------------------
__device__ __forceinline__ v16h load_frag_f32(const float* p, int ld) {
  int lane = threadIdx.x & 31;
  const float* q = p + (size_t)(lane & 15) * ld + ((lane >> 4) << 3);
  v16h f;
#pragma unroll
  for (int h = 0; h < 8; ++h) f[h] = (_Float16)q[h];
#pragma unroll
  for (int h = 0; h < 8; ++h) f[h + 8] = (_Float16)q[h + 16];
  return f;
}

__device__ __forceinline__ v16h load_frag_f16(const _Float16* p, int ld) {
  int lane = threadIdx.x & 31;
  const _Float16* q = p + (size_t)(lane & 15) * ld + ((lane >> 4) << 3);
  v16h f;
#pragma unroll
  for (int h = 0; h < 8; ++h) f[h] = q[h];
#pragma unroll
  for (int h = 0; h < 8; ++h) f[h + 8] = q[h + 16];
  return f;
}

__device__ __forceinline__ v8f wmma_f16(v16h a, v16h b, v8f c) {
  return __builtin_amdgcn_wmma_f32_16x16x32_f16(false, a, false, b, (short)0, c,
                                                false, false);
}

// ---------------------------------------------------------------------------
// Generic GEMM, register-blocked 1x4 N-tiles per wave:
//   C[M x N] = act( A[M x K](f32) * W[Nw x K]^T(f16) + bias[N] )
// Each wave computes a 16x64 strip: A fragment loaded/converted ONCE per
// K-step and reused by 4 consecutive WMMAs. W must have >= Ngroups*64 rows
// (pad with zeros if N not a multiple of 64); stores guarded by Nvalid.
// grid.x = M/16 tiles, grid.y covers ceil(Ngroups/4), block = 128 (4 waves).
// act: 0 = none, 1 = sigmoid, 2 = tanh
// ---------------------------------------------------------------------------
__global__ void gemm_awt_kernel(const float* __restrict__ A, int lda,
                                const _Float16* __restrict__ W, int ldw,
                                const float* __restrict__ bias,
                                float* __restrict__ C, int ldc,
                                int Ngroups, int Nvalid, int K, int act) {
  int wave = threadIdx.x >> 5;
  int mt = blockIdx.x;
  int ng = blockIdx.y * 4 + wave;           // group of 4 consecutive N tiles
  if (ng >= Ngroups) return;                // wave-uniform: EXEC stays all-ones
  int nt0 = ng * 4;
  const float*    Ap = A + (size_t)mt * 16 * lda;
  const _Float16* W0 = W + (size_t)(nt0 + 0) * 16 * ldw;
  const _Float16* W1 = W + (size_t)(nt0 + 1) * 16 * ldw;
  const _Float16* W2 = W + (size_t)(nt0 + 2) * 16 * ldw;
  const _Float16* W3 = W + (size_t)(nt0 + 3) * 16 * ldw;
  v8f acc0 = {}, acc1 = {}, acc2 = {}, acc3 = {};
  for (int k = 0; k < K; k += 32) {
    v16h a  = load_frag_f32(Ap + k, lda);   // converted once, used 4x
    v16h b0 = load_frag_f16(W0 + k, ldw);
    v16h b1 = load_frag_f16(W1 + k, ldw);
    v16h b2 = load_frag_f16(W2 + k, ldw);
    v16h b3 = load_frag_f16(W3 + k, ldw);
    acc0 = wmma_f16(a, b0, acc0);
    acc1 = wmma_f16(a, b1, acc1);
    acc2 = wmma_f16(a, b2, acc2);
    acc3 = wmma_f16(a, b3, acc3);
  }
  int lane = threadIdx.x & 31;
  int mb = mt * 16 + ((lane >> 4) << 3);
  v8f accs[4] = {acc0, acc1, acc2, acc3};
#pragma unroll
  for (int j = 0; j < 4; ++j) {
    int n = (nt0 + j) * 16 + (lane & 15);
    if (n < Nvalid) {                       // tile-uniform for N % 16 == 0
      float bv = bias[n];
#pragma unroll
      for (int r = 0; r < 8; ++r) {
        float v = accs[j][r] + bv;
        if (act == 1)      v = 1.f / (1.f + __expf(-v));
        else if (act == 2) v = tanhf(v);
        C[(size_t)(mb + r) * ldc + n] = v;
      }
    }
  }
}

// a_sum = h_a*W_ha^T + b_ha + h_f*W_hf^T + b_hf   (M=64, N=512, K=512)
// h_a / h_f live inside cat_hat (lda = 4096) at column offsets 1024 / 1536.
__global__ void att_presum_kernel(const float* __restrict__ cat_hat,
                                  const _Float16* __restrict__ wha,
                                  const _Float16* __restrict__ whf,
                                  const float* __restrict__ b_ha,
                                  const float* __restrict__ b_hf,
                                  float* __restrict__ a_sum) {
  int wave = threadIdx.x >> 5;
  int mt = blockIdx.x;
  int nt = blockIdx.y * 4 + wave;
  const float*    Aa = cat_hat + (size_t)mt * 16 * 4096 + 1024;
  const float*    Af = cat_hat + (size_t)mt * 16 * 4096 + 1536;
  const _Float16* Wa = wha + (size_t)nt * 16 * H_;
  const _Float16* Wf = whf + (size_t)nt * 16 * H_;
  v8f acc = {};
  for (int k = 0; k < H_; k += 32) {
    acc = wmma_f16(load_frag_f32(Aa + k, 4096), load_frag_f16(Wa + k, H_), acc);
    acc = wmma_f16(load_frag_f32(Af + k, 4096), load_frag_f16(Wf + k, H_), acc);
  }
  int lane = threadIdx.x & 31;
  int n  = nt * 16 + (lane & 15);
  int mb = mt * 16 + ((lane >> 4) << 3);
  float bv = b_ha[n] + b_hf[n];
#pragma unroll
  for (int r = 0; r < 8; ++r) a_sum[(size_t)(mb + r) * H_ + n] = acc[r] + bv;
}

// ---------------------------------------------------------------------------
// Elementwise / utility kernels
// ---------------------------------------------------------------------------
__global__ void cvt_f16_kernel(const float* __restrict__ s,
                               _Float16* __restrict__ d, int n) {
  int i = blockIdx.x * blockDim.x + threadIdx.x;
  if (i < n) d[i] = (_Float16)s[i];
}

__global__ void zero_f16_kernel(_Float16* __restrict__ p, int n) {
  int i = blockIdx.x * blockDim.x + threadIdx.x;
  if (i < n) p[i] = (_Float16)0.f;
}

__global__ void zero_kernel(float* __restrict__ p, int n) {
  int i = blockIdx.x * blockDim.x + threadIdx.x;
  if (i < n) p[i] = 0.f;
}

__global__ void att_mean_kernel(const float* __restrict__ att_feats,
                                float* __restrict__ att_mean) {
  int idx = blockIdx.x * blockDim.x + threadIdx.x;   // B*AF
  if (idx >= B_ * AF_) return;
  int b = idx / AF_, d = idx % AF_;
  float s = 0.f;
  const float* p = att_feats + ((size_t)b * ATT_) * AF_ + d;
  for (int a = 0; a < ATT_; ++a) s += p[(size_t)a * AF_];
  att_mean[idx] = s * (1.f / (float)ATT_);
}

__global__ void copy_attmean_kernel(const float* __restrict__ att_mean,
                                    float* __restrict__ cat_hat) {
  int idx = blockIdx.x * blockDim.x + threadIdx.x;   // B*AF
  if (idx >= B_ * AF_) return;
  int b = idx / AF_, d = idx % AF_;
  cat_hat[(size_t)b * 4096 + 2048 + d] = att_mean[idx];
}

// Build cat_hat[b] = [x_a | x_f | h_a | h_f | (att_res filled later)],
// cat_f1 = [x_a|h_a], cat_f2 = [x_f|h_f], plus c_a, c_f.
__global__ void gather_kernel(int i, int use_f,
                              const int* __restrict__ word_idx,
                              const int* __restrict__ father_idx,
                              const float* __restrict__ fc_emb,
                              const float* __restrict__ embed_tab,
                              const float* __restrict__ h_states,
                              const float* __restrict__ c_states,
                              float* __restrict__ cat_hat,
                              float* __restrict__ cat_f1,
                              float* __restrict__ cat_f2,
                              float* __restrict__ c_a,
                              float* __restrict__ c_f) {
  int idx = blockIdx.x * blockDim.x + threadIdx.x;   // B*H
  if (idx >= B_ * H_) return;
  int b = idx >> 9, e = idx & (H_ - 1);
  float xa, ha, ca;
  if (i == 0) {
    xa = fc_emb[idx]; ha = 0.f; ca = 0.f;
  } else {
    int f = father_idx[b * T_ + i];
    int w = word_idx[b * T_ + f];
    xa = embed_tab[(size_t)w * E_ + e];
    ha = h_states[((size_t)b * T_ + f) * H_ + e];
    ca = c_states[((size_t)b * T_ + f) * H_ + e];
  }
  float xf = 0.f, hf = 0.f, cf = 0.f;
  if (use_f) {
    int ip = i - 1;
    int w  = word_idx[b * T_ + ip];
    xf = embed_tab[(size_t)w * E_ + e];
    hf = h_states[((size_t)b * T_ + ip) * H_ + e];
    cf = c_states[((size_t)b * T_ + ip) * H_ + e];
  }
  float* ch = cat_hat + (size_t)b * 4096;
  ch[e] = xa; ch[512 + e] = xf; ch[1024 + e] = ha; ch[1536 + e] = hf;
  cat_f1[(size_t)b * 1024 + e] = xa; cat_f1[(size_t)b * 1024 + 512 + e] = ha;
  cat_f2[(size_t)b * 1024 + e] = xf; cat_f2[(size_t)b * 1024 + 512 + e] = hf;
  c_a[idx] = ca; c_f[idx] = cf;
}

// logits[b,a] = sum_e tanh(a_sum[b,e] + p_att[b,a,e]) * W_alpha[e] + b_alpha.
// One wave per (b,a); 8 waves per block.
__global__ void att_logit_kernel(const float* __restrict__ a_sum,
                                 const float* __restrict__ p_att,
                                 const float* __restrict__ W_alpha,
                                 const float* __restrict__ b_alpha,
                                 float* __restrict__ logits) {
  int wid = blockIdx.x * (blockDim.x >> 5) + (threadIdx.x >> 5);
  if (wid >= B_ * ATT_) return;
  int b = wid / ATT_;
  int lane = threadIdx.x & 31;
  const float* pa = p_att + (size_t)wid * H_;
  const float* as = a_sum + (size_t)b * H_;
  float s = 0.f;
  for (int e = lane; e < H_; e += 32)
    s += tanhf(as[e] + pa[e]) * W_alpha[e];
#pragma unroll
  for (int off = 16; off > 0; off >>= 1) s += __shfl_xor(s, off, 32);
  if (lane == 0) logits[wid] = s + b_alpha[0];
}

// softmax over 196 + weighted pooling of att_feats -> cat_hat[:, 2048:4096]
__global__ void att_apply_kernel(const float* __restrict__ logits,
                                 const float* __restrict__ att_feats,
                                 float* __restrict__ cat_hat) {
  __shared__ float w[ATT_];
  __shared__ float red[256];
  int b = blockIdx.x, t = threadIdx.x;
  const float* lg = logits + (size_t)b * ATT_;
  float m = -1e30f;
  for (int a = t; a < ATT_; a += 256) m = fmaxf(m, lg[a]);
  red[t] = m; __syncthreads();
  for (int s = 128; s > 0; s >>= 1) { if (t < s) red[t] = fmaxf(red[t], red[t + s]); __syncthreads(); }
  m = red[0]; __syncthreads();
  float sum = 0.f;
  for (int a = t; a < ATT_; a += 256) { float e = __expf(lg[a] - m); w[a] = e; sum += e; }
  red[t] = sum; __syncthreads();
  for (int s = 128; s > 0; s >>= 1) { if (t < s) red[t] += red[t + s]; __syncthreads(); }
  float inv = 1.f / red[0];
  __syncthreads();
  for (int d = t; d < AF_; d += 256) {
    const float* af = att_feats + ((size_t)b * ATT_) * AF_ + d;
    float acc = 0.f;
    for (int a = 0; a < ATT_; ++a) acc += w[a] * af[(size_t)a * AF_];
    cat_hat[(size_t)b * 4096 + 2048 + d] = acc * inv;
  }
}

// nc = f1*c_a + f2*c_f + in*cellgate ; nh = out*tanh(nc); write slot i.
__global__ void cell_update_kernel(int i,
                                   const float* __restrict__ gates,
                                   const float* __restrict__ cellgate,
                                   const float* __restrict__ f1,
                                   const float* __restrict__ f2,
                                   const float* __restrict__ c_a,
                                   const float* __restrict__ c_f,
                                   float* __restrict__ h_states,
                                   float* __restrict__ c_states) {
  int idx = blockIdx.x * blockDim.x + threadIdx.x;   // B*H
  if (idx >= B_ * H_) return;
  int b = idx >> 9, e = idx & (H_ - 1);
  float ig = gates[(size_t)b * 1024 + e];
  float og = gates[(size_t)b * 1024 + 512 + e];
  float nc = f1[idx] * c_a[idx] + f2[idx] * c_f[idx] + ig * cellgate[idx];
  float nh = og * tanhf(nc);
  h_states[((size_t)b * T_ + i) * H_ + e] = nh;
  c_states[((size_t)b * T_ + i) * H_ + e] = nc;
}

// In-place row-wise log_softmax over V (one block per row of d_out).
__global__ void logsoftmax_kernel(float* __restrict__ out) {
  __shared__ float red[256];
  int row = blockIdx.x, t = threadIdx.x;
  float* p = out + (size_t)row * V_;
  float m = -1e30f;
  for (int j = t; j < V_; j += 256) m = fmaxf(m, p[j]);
  red[t] = m; __syncthreads();
  for (int s = 128; s > 0; s >>= 1) { if (t < s) red[t] = fmaxf(red[t], red[t + s]); __syncthreads(); }
  m = red[0]; __syncthreads();
  float sum = 0.f;
  for (int j = t; j < V_; j += 256) sum += __expf(p[j] - m);
  red[t] = sum; __syncthreads();
  for (int s = 128; s > 0; s >>= 1) { if (t < s) red[t] += red[t + s]; __syncthreads(); }
  float lse = m + __logf(red[0]);
  for (int j = t; j < V_; j += 256) p[j] = p[j] - lse;
}

// ---------------------------------------------------------------------------
extern "C" void kernel_launch(void* const* d_in, const int* in_sizes, int n_in,
                              void* d_out, int out_size, void* d_ws, size_t ws_size,
                              hipStream_t stream) {
  const int*   word_idx   = (const int*)d_in[0];
  const int*   father_idx = (const int*)d_in[1];
  const float* fc_feats   = (const float*)d_in[2];
  const float* att_feats  = (const float*)d_in[3];
  const float* W_fc   = (const float*)d_in[4];
  const float* b_fc   = (const float*)d_in[5];
  const float* W_att  = (const float*)d_in[6];
  const float* b_att  = (const float*)d_in[7];
  const float* embed  = (const float*)d_in[8];
  const float* W_f1   = (const float*)d_in[9];
  const float* b_f1   = (const float*)d_in[10];
  const float* W_f2   = (const float*)d_in[11];
  const float* b_f2   = (const float*)d_in[12];
  const float* W_H    = (const float*)d_in[13];
  const float* b_H    = (const float*)d_in[14];
  const float* W_I    = (const float*)d_in[15];
  const float* b_I    = (const float*)d_in[16];
  const float* W_ha   = (const float*)d_in[17];
  const float* b_ha   = (const float*)d_in[18];
  const float* W_hf   = (const float*)d_in[19];
  const float* b_hf   = (const float*)d_in[20];
  const float* W_al   = (const float*)d_in[21];
  const float* b_al   = (const float*)d_in[22];
  const float* W_lg   = (const float*)d_in[23];
  const float* b_lg   = (const float*)d_in[24];
  float* out = (float*)d_out;

  // workspace carve-out
  char*  ws  = (char*)d_ws;
  size_t off = 0;
  auto alloc = [&](size_t bytes) -> void* {
    off = (off + 255) & ~(size_t)255;
    void* p = ws + off;
    off += bytes;
    return p;
  };
  _Float16* wH16  = (_Float16*)alloc((size_t)1024 * 2048 * 2);
  _Float16* wI16  = (_Float16*)alloc((size_t)512 * 4096 * 2);
  _Float16* wf116 = (_Float16*)alloc((size_t)512 * 1024 * 2);
  _Float16* wf216 = (_Float16*)alloc((size_t)512 * 1024 * 2);
  _Float16* wha16 = (_Float16*)alloc((size_t)512 * 512 * 2);
  _Float16* whf16 = (_Float16*)alloc((size_t)512 * 512 * 2);
  _Float16* watt16 = (_Float16*)alloc((size_t)512 * 2048 * 2);
  _Float16* wfc16  = (_Float16*)alloc((size_t)512 * 2048 * 2);
  _Float16* wlg16  = (_Float16*)alloc((size_t)VPAD_ * 512 * 2);  // padded rows
  float* p_att   = (float*)alloc((size_t)B_ * ATT_ * H_ * 4);
  float* fc_emb  = (float*)alloc((size_t)B_ * H_ * 4);
  float* attmean = (float*)alloc((size_t)B_ * AF_ * 4);
  float* h_st    = (float*)alloc((size_t)B_ * T_ * H_ * 4);
  float* c_st    = (float*)alloc((size_t)B_ * T_ * H_ * 4);
  float* cat_hat = (float*)alloc((size_t)B_ * 4096 * 4);
  float* cat_f1  = (float*)alloc((size_t)B_ * 1024 * 4);
  float* cat_f2  = (float*)alloc((size_t)B_ * 1024 * 4);
  float* c_a     = (float*)alloc((size_t)B_ * H_ * 4);
  float* c_f     = (float*)alloc((size_t)B_ * H_ * 4);
  float* a_sum   = (float*)alloc((size_t)B_ * H_ * 4);
  float* alog    = (float*)alloc((size_t)B_ * ATT_ * 4);
  float* gates   = (float*)alloc((size_t)B_ * 1024 * 4);
  float* cellg   = (float*)alloc((size_t)B_ * H_ * 4);
  float* f1b     = (float*)alloc((size_t)B_ * H_ * 4);
  float* f2b     = (float*)alloc((size_t)B_ * H_ * 4);

  auto cvt = [&](const float* s, _Float16* d, int n) {
    cvt_f16_kernel<<<(n + 255) / 256, 256, 0, stream>>>(s, d, n);
  };
  cvt(W_H, wH16, 1024 * 2048);
  cvt(W_I, wI16, 512 * 4096);
  cvt(W_f1, wf116, 512 * 1024);
  cvt(W_f2, wf216, 512 * 1024);
  cvt(W_ha, wha16, 512 * 512);
  cvt(W_hf, whf16, 512 * 512);
  cvt(W_att, watt16, 512 * 2048);
  cvt(W_fc, wfc16, 512 * 2048);
  cvt(W_lg, wlg16, V_ * 512);
  {  // zero the padded tail rows of W_logit (rows V_..VPAD_-1)
    int n = (VPAD_ - V_) * 512;
    zero_f16_kernel<<<(n + 255) / 256, 256, 0, stream>>>(wlg16 + (size_t)V_ * 512, n);
  }

  // zero h/c state (gathers may read not-yet-written slots)
  {
    int n = B_ * T_ * H_;
    zero_kernel<<<(n + 255) / 256, 256, 0, stream>>>(h_st, n);
    zero_kernel<<<(n + 255) / 256, 256, 0, stream>>>(c_st, n);
  }

  // fc_emb = fc_feats @ W_fc^T + b_fc    (64 x 512 x 2048); Ngroups=8
  gemm_awt_kernel<<<dim3(B_ / 16, 2), 128, 0, stream>>>(
      fc_feats, FC_, wfc16, FC_, b_fc, fc_emb, H_, 8, 512, FC_, 0);
  // p_att = att_feats @ W_att^T + b_att  (12544 x 512 x 2048); Ngroups=8
  gemm_awt_kernel<<<dim3(B_ * ATT_ / 16, 2), 128, 0, stream>>>(
      att_feats, AF_, watt16, AF_, b_att, p_att, H_, 8, 512, AF_, 0);
  att_mean_kernel<<<(B_ * AF_ + 255) / 256, 256, 0, stream>>>(att_feats, attmean);

  const int ebl = (B_ * H_ + 255) / 256;
  for (int i = 0; i < T_; ++i) {
    int use_f = !(i == 0 || ((i - 1) % 3 == 0));
    gather_kernel<<<ebl, 256, 0, stream>>>(i, use_f, word_idx, father_idx,
                                           fc_emb, embed, h_st, c_st, cat_hat,
                                           cat_f1, cat_f2, c_a, c_f);
    if (i == 0) {
      copy_attmean_kernel<<<(B_ * AF_ + 255) / 256, 256, 0, stream>>>(attmean, cat_hat);
    } else {
      att_presum_kernel<<<dim3(B_ / 16, 512 / 16 / 4), 128, 0, stream>>>(
          cat_hat, wha16, whf16, b_ha, b_hf, a_sum);
      att_logit_kernel<<<(B_ * ATT_ + 7) / 8, 256, 0, stream>>>(a_sum, p_att,
                                                                W_al, b_al, alog);
      att_apply_kernel<<<B_, 256, 0, stream>>>(alog, att_feats, cat_hat);
    }
    // gates (sigmoid, N=1024, Ngroups=16), cellgate (tanh), f1/f2 (sigmoid)
    gemm_awt_kernel<<<dim3(B_ / 16, 4), 128, 0, stream>>>(
        cat_hat, 4096, wH16, 2048, b_H, gates, 1024, 16, 1024, 2048, 1);
    gemm_awt_kernel<<<dim3(B_ / 16, 2), 128, 0, stream>>>(
        cat_hat, 4096, wI16, 4096, b_I, cellg, H_, 8, 512, 4096, 2);
    gemm_awt_kernel<<<dim3(B_ / 16, 2), 128, 0, stream>>>(
        cat_f1, 1024, wf116, 1024, b_f1, f1b, H_, 8, 512, 1024, 1);
    gemm_awt_kernel<<<dim3(B_ / 16, 2), 128, 0, stream>>>(
        cat_f2, 1024, wf216, 1024, b_f2, f2b, H_, 8, 512, 1024, 1);
    cell_update_kernel<<<ebl, 256, 0, stream>>>(i, gates, cellg, f1b, f2b,
                                                c_a, c_f, h_st, c_st);
  }

  // logits = h_states(3072 x 512) @ W_logit^T + b_logit -> d_out, log_softmax
  // Ngroups = VPAD_/64 = 157; stores guarded by Nvalid = 10000.
  gemm_awt_kernel<<<dim3(B_ * T_ / 16, (VPAD_ / 64 + 3) / 4), 128, 0, stream>>>(
      h_st, H_, wlg16, H_, b_lg, out, V_, VPAD_ / 64, V_, H_, 0);
  logsoftmax_kernel<<<B_ * T_, 256, 0, stream>>>(out);
}
